// StandardAttentionLayer_17016660427274
// MI455X (gfx1250) — compile-verified, matrix operand order checked by reading, not run on hardware
//
#include <hip/hip_runtime.h>

// ---------------------------------------------------------------------------
// MI455X (gfx1250) transformer layer, WMMA f16 path + async-to-LDS staging.
// B=4 S=2048 E=1024 H=16 Dh=64 FFN=4096; M = B*S = 8192 rows everywhere.
// Compute-bound (~275 GFLOP vs ~5us of HBM traffic at 23.3 TB/s) -> all
// matmuls via v_wmma_f32_16x16x32_f16 (fp32 accum), LDS-staged tiles,
// copy-through tiles staged with GLOBAL_LOAD_ASYNC_TO_LDS_B128 (ASYNCcnt).
// ---------------------------------------------------------------------------

typedef __attribute__((ext_vector_type(16))) _Float16 v16h;
typedef __attribute__((ext_vector_type(8)))  _Float16 v8h;
typedef __attribute__((ext_vector_type(8)))  float    v8f;

__device__ __forceinline__ v8f wmma_f16(v16h a, v16h b, v8f c) {
  // 8 args: (neg_a, A, neg_b, B, c_mod, C, reuse_a, reuse_b)
  return __builtin_amdgcn_wmma_f32_16x16x32_f16(false, a, false, b, (short)0, c,
                                                false, false);
}

// Async DMA 16 bytes/lane global -> LDS (VFLAT GLOBAL_LOAD_ASYNC_TO_LDS_B128,
// tracked on ASYNCcnt). VDST carries the per-lane LDS byte address: the low
// 32 bits of a generic shared pointer ARE the LDS offset (ISA 10.2 aperture).
__device__ __forceinline__ void async_ld_b128(_Float16* lds_dst,
                                              const _Float16* gsrc) {
  unsigned int loff = (unsigned int)(unsigned long long)(void*)lds_dst;
  asm volatile("global_load_async_to_lds_b128 %0, %1, off"
               :: "v"(loff), "v"((unsigned long long)(const void*)gsrc)
               : "memory");
}
__device__ __forceinline__ void wait_async0() {
  asm volatile("s_wait_asynccnt 0" ::: "memory");
}

// Load a 16x32 f16 fragment (A layout; also used for B via an n-major tile).
// Per ISA 7.12.2: lanes 0-15 / 16-31 both hold rows 0-15; lane>=16 adds
// k-base 8; halves [0..7] = k..k+7, halves [8..15] = k+16..k+23.
// Rows must be contiguous with leading dimension `ld` (halves), 16B-aligned.
__device__ __forceinline__ v16h load_frag(const _Float16* p, int ld) {
  const int lane = threadIdx.x & 31;
  const _Float16* q = p + (lane & 15) * ld + ((lane >> 4) * 8);
  union { v16h v; v8h h[2]; } u;
  u.h[0] = *(const v8h*)(q);
  u.h[1] = *(const v8h*)(q + 16);
  return u.v;
}

// ---------------------------------------------------------------------------
// f32 -> f16 conversion (weights)
// ---------------------------------------------------------------------------
__global__ void cvt_f32_f16_kernel(const float* __restrict__ in,
                                   _Float16* __restrict__ out, int n) {
  int i = blockIdx.x * 256 + threadIdx.x;
  if (i < n) out[i] = (_Float16)in[i];
}

// ---------------------------------------------------------------------------
// LayerNorm over 1024 cols, f32 in -> f16 out. One row per 256-thread block.
// ---------------------------------------------------------------------------
__global__ __launch_bounds__(256)
void layernorm_f16_kernel(const float* __restrict__ x, const float* __restrict__ g,
                          const float* __restrict__ b, _Float16* __restrict__ out) {
  const int row = blockIdx.x;
  const int t = threadIdx.x;
  const float* xr = x + (size_t)row * 1024;
  float4 v = ((const float4*)xr)[t];                 // 256 * 4 = 1024
  float s  = v.x + v.y + v.z + v.w;
  float ss = v.x * v.x + v.y * v.y + v.z * v.z + v.w * v.w;
  for (int o = 16; o >= 1; o >>= 1) {
    s  += __shfl_xor(s,  o, 32);
    ss += __shfl_xor(ss, o, 32);
  }
  __shared__ float red0[8], red1[8];
  const int w = t >> 5, lane = t & 31;
  if (lane == 0) { red0[w] = s; red1[w] = ss; }
  __syncthreads();
  float tot = 0.f, tot2 = 0.f;
#pragma unroll
  for (int i = 0; i < 8; ++i) { tot += red0[i]; tot2 += red1[i]; }
  const float mu   = tot * (1.f / 1024.f);
  const float var  = tot2 * (1.f / 1024.f) - mu * mu;
  const float rinv = rsqrtf(var + 1e-5f);
  float4 gg = ((const float4*)g)[t];
  float4 bb = ((const float4*)b)[t];
  _Float16* o = out + (size_t)row * 1024 + t * 4;
  o[0] = (_Float16)((v.x - mu) * rinv * gg.x + bb.x);
  o[1] = (_Float16)((v.y - mu) * rinv * gg.y + bb.y);
  o[2] = (_Float16)((v.z - mu) * rinv * gg.z + bb.z);
  o[3] = (_Float16)((v.w - mu) * rinv * gg.w + bb.w);
}

// ---------------------------------------------------------------------------
// Tiled WMMA GEMM: C[M,N] = A[M,K] * B[K,N], A/B f16 row-major.
// Block = 256 threads (8 waves), block tile 64x128, wave tile 32x32,
// stage depth K=64 -> 8 WMMAs per wave between barriers.
// A staged via async-to-LDS DMA; B transposed into n-major LDS.
// Epilogue: optional bias (f32), exact GELU, f32 residual; f16 or f32 out.
// ---------------------------------------------------------------------------
template <bool HAS_BIAS, bool GELU, bool HAS_RES, bool OUT16>
__global__ __launch_bounds__(256)
void gemm_f16_kernel(const _Float16* __restrict__ A, const _Float16* __restrict__ B,
                     const float* __restrict__ bias, const float* __restrict__ res,
                     _Float16* __restrict__ out16, float* __restrict__ out32,
                     int M, int N, int K) {
  __shared__ _Float16 sA[64 * 72];    // [row][k]  row-major, ld 72 (144B rows)
  __shared__ _Float16 sB[128 * 72];   // [n][k]    n-major (transposed on store)
  const int t = threadIdx.x;
  const int lane = t & 31;
  const int w = t >> 5;
  const int waveM = w & 1;            // 2 waves along M
  const int waveN = w >> 1;           // 4 waves along N
  const int bm = blockIdx.y * 64;
  const int bn = blockIdx.x * 128;

  v8f acc[2][2] = {};

  const int ar = t >> 2, ac = (t & 3) * 16;   // A stage: 64 rows x 64 halves
  const int br = t >> 2, bc = (t & 3) * 32;   // B stage: 64 k-rows x 128 n

  for (int kk = 0; kk < K; kk += 64) {
    __syncthreads();
    // --- stage A tile: async DMA global->LDS (2 x b128 per lane) ---
    const _Float16* ap = A + (size_t)(bm + ar) * K + kk + ac;
    async_ld_b128(sA + ar * 72 + ac, ap);
    async_ld_b128(sA + ar * 72 + ac + 8, ap + 8);
    // --- stage B tile transposed into n-major LDS ---
    const _Float16* bp = B + (size_t)(kk + br) * N + bn + bc;
    v8h bv0 = *(const v8h*)(bp);
    v8h bv1 = *(const v8h*)(bp + 8);
    v8h bv2 = *(const v8h*)(bp + 16);
    v8h bv3 = *(const v8h*)(bp + 24);
#pragma unroll
    for (int i = 0; i < 8; ++i) {
      sB[(bc + i) * 72 + br]      = bv0[i];
      sB[(bc + 8 + i) * 72 + br]  = bv1[i];
      sB[(bc + 16 + i) * 72 + br] = bv2[i];
      sB[(bc + 24 + i) * 72 + br] = bv3[i];
    }
    if (kk + 64 < K) {  // global_prefetch_b8 on next tiles
      __builtin_prefetch(A + (size_t)(bm + ar) * K + kk + 64 + ac, 0, 3);
      __builtin_prefetch(B + (size_t)(kk + 64 + br) * N + bn + bc, 0, 3);
    }
    wait_async0();      // drain ASYNCcnt before the barrier (asm not tracked)
    __syncthreads();
#pragma unroll
    for (int kf = 0; kf < 2; ++kf) {
      v16h a0 = load_frag(sA + (waveM * 32) * 72 + kf * 32, 72);
      v16h a1 = load_frag(sA + (waveM * 32 + 16) * 72 + kf * 32, 72);
      v16h b0 = load_frag(sB + (waveN * 32) * 72 + kf * 32, 72);
      v16h b1 = load_frag(sB + (waveN * 32 + 16) * 72 + kf * 32, 72);
      acc[0][0] = wmma_f16(a0, b0, acc[0][0]);
      acc[0][1] = wmma_f16(a0, b1, acc[0][1]);
      acc[1][0] = wmma_f16(a1, b0, acc[1][0]);
      acc[1][1] = wmma_f16(a1, b1, acc[1][1]);
    }
  }

  // Epilogue. C layout: lanes 0-15 rows 0-7, lanes 16-31 rows 8-15; col=lane&15.
  const int colBase = bn + waveN * 32 + (lane & 15);
  const int rowBase = bm + waveM * 32 + ((lane >> 4) * 8);
#pragma unroll
  for (int mi = 0; mi < 2; ++mi)
#pragma unroll
    for (int ni = 0; ni < 2; ++ni) {
      const int col = colBase + ni * 16;
      const float bv = HAS_BIAS ? bias[col] : 0.f;
#pragma unroll
      for (int e = 0; e < 8; ++e) {
        const int row = rowBase + mi * 16 + e;
        float r = acc[mi][ni][e];
        if (HAS_BIAS) r += bv;
        if (GELU) r = 0.5f * r * (1.f + erff(r * 0.70710678118654752f));
        if (HAS_RES) r += res[(size_t)row * N + col];
        if (OUT16) out16[(size_t)row * N + col] = (_Float16)r;
        else       out32[(size_t)row * N + col] = r;
      }
    }
}

// ---------------------------------------------------------------------------
// Causal flash attention. Q/K/V/O are f16 [B, S, H*Dh] (head at col h*64).
// Block = 128 threads (4 waves); block owns (b, h, 64 queries); each wave
// owns 16 queries. Key blocks of 32; scores and P@V both via WMMA.
// K tile staged via async-to-LDS DMA (row-major K already IS the Kt
// B-fragment layout); V transposed into LDS.
// ---------------------------------------------------------------------------
__global__ __launch_bounds__(128)
void attn_kernel(const _Float16* __restrict__ Q, const _Float16* __restrict__ K,
                 const _Float16* __restrict__ V, _Float16* __restrict__ O) {
  __shared__ _Float16 sK[32 * 72];     // [key][dim], ld 72 (144B rows)
  __shared__ _Float16 sVt[64 * 40];    // [dim][key] transposed
  __shared__ _Float16 sP[4][16 * 40];  // per-wave P tile [qrow][key]

  const int t = threadIdx.x, lane = t & 31, w = t >> 5;
  const int b = blockIdx.z, h = blockIdx.y;
  const int qblk = blockIdx.x * 64;
  const int q0 = qblk + w * 16;

  const _Float16* qbase = Q + ((size_t)(b * 2048) + q0) * 1024 + h * 64;
  const v16h qa0 = load_frag(qbase, 1024);       // dims 0..31
  const v16h qa1 = load_frag(qbase + 32, 1024);  // dims 32..63

  v8f o[4] = {};
  float mrun[8], lrun[8];
#pragma unroll
  for (int e = 0; e < 8; ++e) { mrun[e] = -1e30f; lrun[e] = 0.f; }
  const float sscale = 0.125f;  // 1/sqrt(64)
  const int nkb = qblk / 32 + 2;

  const int kr = t >> 2, cc = (t & 3) * 16;  // staging: 32 keys x 64 dims

  for (int kb = 0; kb < nkb; ++kb) {
    __syncthreads();
    // --- stage K tile via async DMA (row-major == Kt B-fragment layout) ---
    const _Float16* kp = K + ((size_t)(b * 2048) + kb * 32 + kr) * 1024 + h * 64 + cc;
    async_ld_b128(sK + kr * 72 + cc, kp);
    async_ld_b128(sK + kr * 72 + cc + 8, kp + 8);
    // --- stage V transposed ---
    const _Float16* vp = V + ((size_t)(b * 2048) + kb * 32 + kr) * 1024 + h * 64 + cc;
    v8h v0 = *(const v8h*)vp;
    v8h v1 = *(const v8h*)(vp + 8);
#pragma unroll
    for (int i = 0; i < 8; ++i) {
      sVt[(cc + i) * 40 + kr]     = v0[i];
      sVt[(cc + 8 + i) * 40 + kr] = v1[i];
    }
    wait_async0();
    __syncthreads();

    // --- scores S = Q @ K^T (16 queries x 32 keys) ---
    v8f c0 = {}, c1 = {};
    v16h bk;
    bk = load_frag(sK, 72);            c0 = wmma_f16(qa0, bk, c0);
    bk = load_frag(sK + 32, 72);       c0 = wmma_f16(qa1, bk, c0);
    bk = load_frag(sK + 16 * 72, 72);      c1 = wmma_f16(qa0, bk, c1);
    bk = load_frag(sK + 16 * 72 + 32, 72); c1 = wmma_f16(qa1, bk, c1);

    // --- causal mask + online softmax (row reductions in 16-lane groups) ---
    const int key0 = kb * 32 + (lane & 15);
    const int key1 = key0 + 16;
    _Float16* pp = sP[w];
#pragma unroll
    for (int e = 0; e < 8; ++e) {
      const int row = e + ((lane >> 4) * 8);
      const int qi = q0 + row;
      float s0 = (key0 <= qi) ? c0[e] * sscale : -1e30f;
      float s1 = (key1 <= qi) ? c1[e] * sscale : -1e30f;
      float mx = fmaxf(s0, s1);
      mx = fmaxf(mx, __shfl_xor(mx, 1, 32));
      mx = fmaxf(mx, __shfl_xor(mx, 2, 32));
      mx = fmaxf(mx, __shfl_xor(mx, 4, 32));
      mx = fmaxf(mx, __shfl_xor(mx, 8, 32));
      const float mnew  = fmaxf(mrun[e], mx);
      const float alpha = __expf(mrun[e] - mnew);
      mrun[e] = mnew;
      const float p0 = __expf(s0 - mnew);
      const float p1 = __expf(s1 - mnew);
      float rs = p0 + p1;
      rs += __shfl_xor(rs, 1, 32);
      rs += __shfl_xor(rs, 2, 32);
      rs += __shfl_xor(rs, 4, 32);
      rs += __shfl_xor(rs, 8, 32);
      lrun[e] = lrun[e] * alpha + rs;
      o[0][e] *= alpha; o[1][e] *= alpha; o[2][e] *= alpha; o[3][e] *= alpha;
      pp[row * 40 + (lane & 15)]      = (_Float16)p0;
      pp[row * 40 + (lane & 15) + 16] = (_Float16)p1;
    }
    // P written cross-lane within this wave via LDS; drain DS before reading.
    asm volatile("s_wait_dscnt 0" ::: "memory");

    // --- O += P @ V (16 x 32 times 32 x 64) ---
    const v16h pa = load_frag(pp, 40);
#pragma unroll
    for (int j = 0; j < 4; ++j) {
      v16h bv = load_frag(sVt + j * 16 * 40, 40);
      o[j] = wmma_f16(pa, bv, o[j]);
    }
  }

  // --- normalize and write out ---
#pragma unroll
  for (int j = 0; j < 4; ++j)
#pragma unroll
    for (int e = 0; e < 8; ++e) {
      const int row = e + ((lane >> 4) * 8);
      const float r = o[j][e] / lrun[e];
      O[((size_t)(b * 2048) + q0 + row) * 1024 + h * 64 + j * 16 + (lane & 15)] =
          (_Float16)r;
    }
}

// ---------------------------------------------------------------------------
// Host-side orchestration.
// Inputs: x, Wq, Wk, Wv, Wo, ln1_g, ln1_b, ln2_g, ln2_b, W1, b1, W2, b2
// ---------------------------------------------------------------------------
extern "C" void kernel_launch(void* const* d_in, const int* in_sizes, int n_in,
                              void* d_out, int out_size, void* d_ws, size_t ws_size,
                              hipStream_t stream) {
  (void)in_sizes; (void)n_in; (void)out_size; (void)ws_size;
  const float* x    = (const float*)d_in[0];
  const float* Wq   = (const float*)d_in[1];
  const float* Wk   = (const float*)d_in[2];
  const float* Wv   = (const float*)d_in[3];
  const float* Wo   = (const float*)d_in[4];
  const float* ln1g = (const float*)d_in[5];
  const float* ln1b = (const float*)d_in[6];
  const float* ln2g = (const float*)d_in[7];
  const float* ln2b = (const float*)d_in[8];
  const float* W1   = (const float*)d_in[9];
  const float* b1   = (const float*)d_in[10];
  const float* W2   = (const float*)d_in[11];
  const float* b2   = (const float*)d_in[12];
  float* out = (float*)d_out;

  const int M = 8192, E = 1024, F = 4096;
  char* ws = (char*)d_ws;
  const size_t MB = 1024ull * 1024ull;
  _Float16* Wq16 = (_Float16*)(ws + 0 * MB);    // 2 MB
  _Float16* Wk16 = (_Float16*)(ws + 2 * MB);    // 2 MB
  _Float16* Wv16 = (_Float16*)(ws + 4 * MB);    // 2 MB
  _Float16* Wo16 = (_Float16*)(ws + 6 * MB);    // 2 MB
  _Float16* W116 = (_Float16*)(ws + 8 * MB);    // 8 MB
  _Float16* W216 = (_Float16*)(ws + 16 * MB);   // 8 MB
  _Float16* H16  = (_Float16*)(ws + 24 * MB);   // 16 MB (LN1 out, reused for LN2)
  _Float16* Q16  = (_Float16*)(ws + 40 * MB);   // 16 MB
  _Float16* K16  = (_Float16*)(ws + 56 * MB);   // 16 MB
  _Float16* V16  = (_Float16*)(ws + 72 * MB);   // 16 MB
  _Float16* A16  = (_Float16*)(ws + 88 * MB);   // 16 MB (attn out)
  float*    X2   = (float*)   (ws + 104 * MB);  // 32 MB (residual after attn)
  _Float16* G16  = (_Float16*)(ws + 40 * MB);   // 64 MB GELU acts (reuses Q/K/V/A16)

  // 1) weights -> f16
  cvt_f32_f16_kernel<<<(E * E) / 256, 256, 0, stream>>>(Wq, Wq16, E * E);
  cvt_f32_f16_kernel<<<(E * E) / 256, 256, 0, stream>>>(Wk, Wk16, E * E);
  cvt_f32_f16_kernel<<<(E * E) / 256, 256, 0, stream>>>(Wv, Wv16, E * E);
  cvt_f32_f16_kernel<<<(E * E) / 256, 256, 0, stream>>>(Wo, Wo16, E * E);
  cvt_f32_f16_kernel<<<(E * F) / 256, 256, 0, stream>>>(W1, W116, E * F);
  cvt_f32_f16_kernel<<<(F * E) / 256, 256, 0, stream>>>(W2, W216, F * E);

  // 2) LN1
  layernorm_f16_kernel<<<M, 256, 0, stream>>>(x, ln1g, ln1b, H16);

  // 3) Q/K/V projections
  const dim3 gE(E / 128, M / 64);
  gemm_f16_kernel<false, false, false, true><<<gE, 256, 0, stream>>>(
      H16, Wq16, nullptr, nullptr, Q16, nullptr, M, E, E);
  gemm_f16_kernel<false, false, false, true><<<gE, 256, 0, stream>>>(
      H16, Wk16, nullptr, nullptr, K16, nullptr, M, E, E);
  gemm_f16_kernel<false, false, false, true><<<gE, 256, 0, stream>>>(
      H16, Wv16, nullptr, nullptr, V16, nullptr, M, E, E);

  // 4) causal flash attention
  attn_kernel<<<dim3(2048 / 64, 16, 4), 128, 0, stream>>>(Q16, K16, V16, A16);

  // 5) output projection + residual(x) -> X2 (f32)
  gemm_f16_kernel<false, false, true, false><<<gE, 256, 0, stream>>>(
      A16, Wo16, nullptr, x, nullptr, X2, M, E, E);

  // 6) LN2
  layernorm_f16_kernel<<<M, 256, 0, stream>>>(X2, ln2g, ln2b, H16);

  // 7) FFN1: h @ W1 + b1 -> GELU -> f16
  gemm_f16_kernel<true, true, false, true><<<dim3(F / 128, M / 64), 256, 0, stream>>>(
      H16, W116, b1, nullptr, G16, nullptr, M, F, E);

  // 8) FFN2: g @ W2 + b2 + residual(X2) -> d_out (f32)
  gemm_f16_kernel<true, false, true, false><<<gE, 256, 0, stream>>>(
      G16, W216, b2, X2, nullptr, out, M, E, F);
}
